// GVAE_64089501991493
// MI455X (gfx1250) — matrix operand order repaired and checked by reading, not compiled, not run on hardware
//
#include <hip/hip_runtime.h>
#include <stdint.h>

// ---------------- problem constants ----------------
#define NNODES 50000
#define NEDGES 800000
#define BN_EPS 1e-5f

// ---------------- WMMA types / helpers ----------------
typedef __attribute__((ext_vector_type(16))) __bf16 v16bf;
typedef __attribute__((ext_vector_type(2)))  __bf16 v2bf;
typedef __attribute__((ext_vector_type(8)))  float  v8f;

union ABReg  { v16bf v; v2bf h2[8]; unsigned short s[16]; uint4 q[2]; };
union AccReg { v8f v; float f[8]; };
union HsReg  { uint4 q; v2bf h2[4]; unsigned short s[8]; };
union HdReg  { uint4 q[2]; unsigned short s[16]; };

// native f32 -> bf16 (backend emits hardware cvt, RNE)
__device__ __forceinline__ unsigned short f2bf(float f) {
  return __builtin_bit_cast(unsigned short, (__bf16)f);
}
__device__ __forceinline__ float bf2f(unsigned short h) {
  unsigned u = ((unsigned)h) << 16;
  return __builtin_bit_cast(float, u);
}
// compiler-level memory fence: DS ops of one wave are executed in order by the
// LDS pipe, so for cross-lane same-wave LDS communication we only need to stop
// the compiler from reordering loads before stores.
#define WAVE_LDS_FENCE() asm volatile("" ::: "memory")

// =====================================================================
// K0: hid[e][c] = relu(edge_attr @ nn_w1 + nn_b1), stored bf16  (25.6 MB)
// =====================================================================
__global__ void k_hid(const float* __restrict__ ea, const float* __restrict__ w1,
                      const float* __restrict__ b1, unsigned short* __restrict__ hid) {
  __shared__ float sw1[8 * 16];
  __shared__ float sb1[16];
  for (int i = threadIdx.x; i < 128; i += blockDim.x) sw1[i] = w1[i];
  if (threadIdx.x < 16) sb1[threadIdx.x] = b1[threadIdx.x];
  __syncthreads();
  for (int e = blockIdx.x * blockDim.x + threadIdx.x; e < NEDGES;
       e += gridDim.x * blockDim.x) {
    float a[8];
    const float* row = ea + (size_t)e * 8;
#pragma unroll
    for (int i = 0; i < 8; ++i) a[i] = row[i];
#pragma unroll
    for (int o = 0; o < 16; ++o) {
      float acc = sb1[o];
#pragma unroll
      for (int i = 0; i < 8; ++i) acc = fmaf(a[i], sw1[i * 16 + o], acc);
      hid[(size_t)e * 16 + o] = f2bf(fmaxf(acc, 0.f));
    }
  }
}

// =====================================================================
// K1: cast f32 -> bf16 (used once for x -> h_bf16)
// =====================================================================
__global__ void k_cast(const float* __restrict__ in, unsigned short* __restrict__ out,
                       int n) {
  for (int i = blockIdx.x * blockDim.x + threadIdx.x; i < n;
       i += gridDim.x * blockDim.x)
    out[i] = f2bf(in[i]);
}

// =====================================================================
// K2: zero agg + BN stats
// =====================================================================
__global__ void k_zero(float* __restrict__ agg, float* __restrict__ stats) {
  int total = NNODES * 16;
  for (int i = blockIdx.x * blockDim.x + threadIdx.x; i < total;
       i += gridDim.x * blockDim.x)
    agg[i] = 0.f;
  if (blockIdx.x == 0 && threadIdx.x < 32) stats[threadIdx.x] = 0.f;
}

// =====================================================================
// K3: fused NNConv message + scatter.
// msg_tile[16x16] = P[16x288] @ B[288x16] via 9 chained bf16 WMMAs, where
//   P[e, c*16+i] = hid[e,c] * h_src[e,i]   (K<256)   <- v_pk_mul_bf16
//   P[e, 256+i]  = h_src[e,i]              (bias block, K 256..271)
//   B[K][o]      = nn_w2[c, i*16+o] / nn_b2[i*16+o] / 0  (LDS, transposed)
// then scatter-add into agg[dst] straight from the C/D register layout.
// =====================================================================
__global__ __launch_bounds__(256) void k_conv(
    const unsigned short* __restrict__ hbf,   // [N][16] bf16 node features
    const unsigned short* __restrict__ hid,   // [E][16] bf16 edge hidden
    const int* __restrict__ eidx, const float* __restrict__ w2,
    const float* __restrict__ b2, float* __restrict__ agg) {
  __shared__ unsigned short W2T[16 * 288];  // [o][k], bf16, padded K=288
  for (int idx = threadIdx.x; idx < 16 * 288; idx += blockDim.x) {
    int o = idx / 288, k = idx - o * 288;
    float v = 0.f;
    if (k < 256)      v = w2[(k >> 4) * 256 + (k & 15) * 16 + o];
    else if (k < 272) v = b2[(k - 256) * 16 + o];
    W2T[idx] = f2bf(v);
  }
  __syncthreads();

  const int lane = threadIdx.x & 31;
  const int wid  = threadIdx.x >> 5;
  const int m    = lane & 15;   // edge-in-tile (A row M) / output column N
  const int hi   = lane >> 4;   // lane half
  const int ntiles = NEDGES / 16;

  for (int tile = blockIdx.x * (blockDim.x >> 5) + wid; tile < ntiles;
       tile += gridDim.x * (blockDim.x >> 5)) {
    const int e   = tile * 16 + m;
    const int src = eidx[e];

    // full hid row (16 bf16), needed by both lane halves
    HdReg hd;
    {
      const uint4* hp = (const uint4*)(hid + (size_t)e * 16);
      hd.q[0] = hp[0];
      hd.q[1] = hp[1];
    }
    // half of h[src] row selected by lane half (i = hi*8 + j), bf16, L2-resident
    HsReg hs;
    hs.q = *(const uint4*)(hbf + (size_t)src * 16 + hi * 8);

    AccReg acc;
#pragma unroll
    for (int r = 0; r < 8; ++r) acc.f[r] = 0.f;

    const unsigned short* brow = &W2T[m * 288];
#pragma unroll
    for (int chunk = 0; chunk < 9; ++chunk) {
      ABReg A, B;
      if (chunk < 8) {
        // A element e2 -> K = chunk*32 + hi*8 + (e2<8 ? e2 : e2+8)
        //   => c = 2*chunk + (e2>=8), i = hi*8 + (e2&7)
        __bf16 c0 = __builtin_bit_cast(__bf16, hd.s[2 * chunk]);
        __bf16 c1 = __builtin_bit_cast(__bf16, hd.s[2 * chunk + 1]);
        v2bf b0; b0[0] = c0; b0[1] = c0;
        v2bf b1; b1[0] = c1; b1[1] = c1;
#pragma unroll
        for (int j = 0; j < 4; ++j) {
          A.h2[j]     = b0 * hs.h2[j];   // v_pk_mul_bf16
          A.h2[4 + j] = b1 * hs.h2[j];
        }
      } else {
        // bias block: A[e, 256+j] = h_src[j] (j<16), 0 otherwise
        A.q[0] = hs.q;
        uint4 z4; z4.x = 0u; z4.y = 0u; z4.z = 0u; z4.w = 0u;
        A.q[1] = z4;
      }
      // B element e2 -> K = chunk*32 + hi*8 + (e2<8 ? e2 : e2+8) (matches A)
      B.q[0] = *(const uint4*)(brow + chunk * 32 + hi * 8);
      B.q[1] = *(const uint4*)(brow + chunk * 32 + hi * 8 + 16);
      acc.v = __builtin_amdgcn_wmma_f32_16x16x32_bf16(
          false, A.v, false, B.v, (short)0, acc.v, false, false);
    }

    // D layout: lane holds msg[M = r + hi*8][N = m]; scatter to agg[dst]
#pragma unroll
    for (int r = 0; r < 8; ++r) {
      int er = tile * 16 + r + hi * 8;
      int d  = eidx[NEDGES + er];
      atomicAdd(&agg[(size_t)d * 16 + m], acc.f[r]);
    }
  }
}

// =====================================================================
// K4: hpre = relu(agg + h@root + cb); accumulate per-channel sum/sumsq
// =====================================================================
__global__ void k_root(const unsigned short* __restrict__ h_in,
                       const float* __restrict__ agg,
                       const float* __restrict__ root, const float* __restrict__ cb,
                       float* __restrict__ hpre, float* __restrict__ stats) {
  __shared__ float sroot[256], scb[16], ssum[32];
  for (int i = threadIdx.x; i < 256; i += blockDim.x) sroot[i] = root[i];
  if (threadIdx.x < 16) scb[threadIdx.x] = cb[threadIdx.x];
  if (threadIdx.x < 32) ssum[threadIdx.x] = 0.f;
  __syncthreads();

  float ls[16], lq[16];
#pragma unroll
  for (int o = 0; o < 16; ++o) { ls[o] = 0.f; lq[o] = 0.f; }

  for (int n = blockIdx.x * blockDim.x + threadIdx.x; n < NNODES;
       n += gridDim.x * blockDim.x) {
    HdReg hr16;
    {
      const uint4* hp = (const uint4*)(h_in + (size_t)n * 16);
      hr16.q[0] = hp[0];
      hr16.q[1] = hp[1];
    }
    float hrow[16];
#pragma unroll
    for (int i = 0; i < 16; ++i) hrow[i] = bf2f(hr16.s[i]);
#pragma unroll
    for (int o = 0; o < 16; ++o) {
      float a = agg[(size_t)n * 16 + o] + scb[o];
#pragma unroll
      for (int i = 0; i < 16; ++i) a = fmaf(hrow[i], sroot[i * 16 + o], a);
      a = fmaxf(a, 0.f);
      hpre[(size_t)n * 16 + o] = a;
      ls[o] += a;
      lq[o] += a * a;
    }
  }
#pragma unroll
  for (int o = 0; o < 16; ++o) {
    atomicAdd(&ssum[o], ls[o]);        // ds_add_f32
    atomicAdd(&ssum[16 + o], lq[o]);
  }
  __syncthreads();
  if (threadIdx.x < 32) atomicAdd(&stats[threadIdx.x], ssum[threadIdx.x]);
}

// =====================================================================
// K5: BatchNorm apply (train-mode biased variance), writes bf16 h
// =====================================================================
__global__ void k_bn(const float* __restrict__ hpre, const float* __restrict__ stats,
                     const float* __restrict__ g, const float* __restrict__ be,
                     unsigned short* __restrict__ hout) {
  __shared__ float sm[16], sv[16], sg[16], sb[16];
  if (threadIdx.x < 16) {
    float mean = stats[threadIdx.x] * (1.f / NNODES);
    float var  = stats[16 + threadIdx.x] * (1.f / NNODES) - mean * mean;
    sm[threadIdx.x] = mean;
    sv[threadIdx.x] = rsqrtf(var + BN_EPS);
    sg[threadIdx.x] = g[threadIdx.x];
    sb[threadIdx.x] = be[threadIdx.x];
  }
  __syncthreads();
  int total = NNODES * 16;
  for (int i = blockIdx.x * blockDim.x + threadIdx.x; i < total;
       i += gridDim.x * blockDim.x) {
    int o = i & 15;
    hout[i] = f2bf(sg[o] * (hpre[i] - sm[o]) * sv[o] + sb[o]);
  }
}

// =====================================================================
// K6: latent: z = mu + eps*exp(0.5*min(logvar,10)), stored bf16 for decoder
// =====================================================================
__global__ void k_latent(const unsigned short* __restrict__ h,
                         const float* __restrict__ eps,
                         const float* __restrict__ mu_w, const float* __restrict__ mu_b,
                         const float* __restrict__ lv_w, const float* __restrict__ lv_b,
                         unsigned short* __restrict__ zbf) {
  __shared__ float smw[256], slw[256], smb[16], slb[16];
  for (int i = threadIdx.x; i < 256; i += blockDim.x) { smw[i] = mu_w[i]; slw[i] = lv_w[i]; }
  if (threadIdx.x < 16) { smb[threadIdx.x] = mu_b[threadIdx.x]; slb[threadIdx.x] = lv_b[threadIdx.x]; }
  __syncthreads();
  for (int n = blockIdx.x * blockDim.x + threadIdx.x; n < NNODES;
       n += gridDim.x * blockDim.x) {
    HdReg hr16;
    {
      const uint4* hp = (const uint4*)(h + (size_t)n * 16);
      hr16.q[0] = hp[0];
      hr16.q[1] = hp[1];
    }
    float hr[16];
#pragma unroll
    for (int i = 0; i < 16; ++i) hr[i] = bf2f(hr16.s[i]);
#pragma unroll
    for (int j = 0; j < 16; ++j) {
      float mu = smb[j], lv = slb[j];
#pragma unroll
      for (int i = 0; i < 16; ++i) {
        mu = fmaf(hr[i], smw[i * 16 + j], mu);
        lv = fmaf(hr[i], slw[i * 16 + j], lv);
      }
      lv = fminf(lv, 10.f);
      float z = mu + eps[(size_t)n * 16 + j] * __expf(0.5f * lv);
      zbf[(size_t)n * 16 + j] = f2bf(z);
    }
  }
}

// =====================================================================
// K7: fused decoder MLP (32->64->64->64->64->8), all layers WMMA.
// Per 16-edge tile: 4 + 3*8 + 2 = 30 v_wmma_f32_16x16x32_bf16.
// =====================================================================
__device__ __forceinline__ void dec_layer64(unsigned short* myact,
                                            const unsigned short* wt,
                                            const float* bs, int m, int hi,
                                            AccReg* acc) {
  WAVE_LDS_FENCE();
  ABReg Ac[2];
#pragma unroll
  for (int c = 0; c < 2; ++c) {
    const unsigned short* ap = &myact[m * 64 + c * 32 + hi * 8];
    Ac[c].q[0] = *(const uint4*)ap;
    Ac[c].q[1] = *(const uint4*)(ap + 16);
  }
#pragma unroll
  for (int t = 0; t < 4; ++t) {
    v8f z = {};
#pragma unroll
    for (int c = 0; c < 2; ++c) {
      ABReg B;
      const unsigned short* bp = &wt[(t * 16 + m) * 64 + c * 32 + hi * 8];
      B.q[0] = *(const uint4*)bp;
      B.q[1] = *(const uint4*)(bp + 16);
      z = __builtin_amdgcn_wmma_f32_16x16x32_bf16(false, Ac[c].v, false, B.v,
                                                  (short)0, z, false, false);
    }
    acc[t].v = z;
  }
  WAVE_LDS_FENCE();
#pragma unroll
  for (int t = 0; t < 4; ++t) {
    float bias = bs[t * 16 + m];
#pragma unroll
    for (int r = 0; r < 8; ++r)
      myact[(r + hi * 8) * 64 + t * 16 + m] =
          f2bf(fmaxf(acc[t].f[r] + bias, 0.f));
  }
}

__global__ __launch_bounds__(256) void k_decoder(
    const unsigned short* __restrict__ zbf, const int* __restrict__ eidx,
    const float* __restrict__ dw0, const float* __restrict__ db0,
    const float* __restrict__ dw1, const float* __restrict__ db1,
    const float* __restrict__ dw2, const float* __restrict__ db2,
    const float* __restrict__ dw3, const float* __restrict__ db3,
    const float* __restrict__ dw4, const float* __restrict__ db4,
    float* __restrict__ out) {
  __shared__ unsigned short w0t[64 * 32];   // [col][k]
  __shared__ unsigned short w1t[64 * 64];
  __shared__ unsigned short w2t[64 * 64];
  __shared__ unsigned short w3t[64 * 64];
  __shared__ unsigned short w4t[16 * 64];   // N padded 8->16
  __shared__ float sb0[64], sb1[64], sb2[64], sb3[64], sb4[16];
  __shared__ unsigned short act[8][16 * 64];  // per-wave activation tile [m][k]

  for (int i = threadIdx.x; i < 64 * 32; i += blockDim.x) {
    int n = i >> 5, k = i & 31;
    w0t[i] = f2bf(dw0[k * 64 + n]);
  }
  for (int i = threadIdx.x; i < 64 * 64; i += blockDim.x) {
    int n = i >> 6, k = i & 63;
    w1t[i] = f2bf(dw1[k * 64 + n]);
    w2t[i] = f2bf(dw2[k * 64 + n]);
    w3t[i] = f2bf(dw3[k * 64 + n]);
  }
  for (int i = threadIdx.x; i < 16 * 64; i += blockDim.x) {
    int n = i >> 6, k = i & 63;
    w4t[i] = (n < 8) ? f2bf(dw4[k * 8 + n]) : (unsigned short)0;
  }
  for (int i = threadIdx.x; i < 64; i += blockDim.x) {
    sb0[i] = db0[i]; sb1[i] = db1[i]; sb2[i] = db2[i]; sb3[i] = db3[i];
  }
  if (threadIdx.x < 16) sb4[threadIdx.x] = (threadIdx.x < 8) ? db4[threadIdx.x] : 0.f;
  __syncthreads();

  const int lane = threadIdx.x & 31;
  const int wid  = threadIdx.x >> 5;
  const int m    = lane & 15;
  const int hi   = lane >> 4;
  unsigned short* myact = act[wid];
  const int ntiles = NEDGES / 16;

  for (int tile = blockIdx.x * (blockDim.x >> 5) + wid; tile < ntiles;
       tile += gridDim.x * (blockDim.x >> 5)) {
    const int e   = tile * 16 + m;
    const int src = eidx[e];
    const int dst = eidx[NEDGES + e];

    // a0 = [z_src | z_dst] directly in A layout (K=32):
    // e2<8 -> K = e2+hi*8 (z_src half), e2>=8 -> K = 16+(e2-8)+hi*8 (z_dst half)
    ABReg A0;
    A0.q[0] = *(const uint4*)(zbf + (size_t)src * 16 + hi * 8);
    A0.q[1] = *(const uint4*)(zbf + (size_t)dst * 16 + hi * 8);

    AccReg acc[4];
    // ---- layer 0: K=32, N=64 (4 tiles) ----
#pragma unroll
    for (int t = 0; t < 4; ++t) {
      ABReg B;
      const unsigned short* bp = &w0t[(t * 16 + m) * 32 + hi * 8];
      B.q[0] = *(const uint4*)bp;
      B.q[1] = *(const uint4*)(bp + 16);
      v8f z = {};
      acc[t].v = __builtin_amdgcn_wmma_f32_16x16x32_bf16(false, A0.v, false, B.v,
                                                         (short)0, z, false, false);
    }
    WAVE_LDS_FENCE();
#pragma unroll
    for (int t = 0; t < 4; ++t) {
      float bias = sb0[t * 16 + m];
#pragma unroll
      for (int r = 0; r < 8; ++r)
        myact[(r + hi * 8) * 64 + t * 16 + m] =
            f2bf(fmaxf(acc[t].f[r] + bias, 0.f));
    }
    // ---- layers 1..3: K=64, N=64 ----
    dec_layer64(myact, w1t, sb1, m, hi, acc);
    dec_layer64(myact, w2t, sb2, m, hi, acc);
    dec_layer64(myact, w3t, sb3, m, hi, acc);

    // ---- layer 4: K=64, N=8 (padded to 16), no relu ----
    WAVE_LDS_FENCE();
    ABReg Ac[2];
#pragma unroll
    for (int c = 0; c < 2; ++c) {
      const unsigned short* ap = &myact[m * 64 + c * 32 + hi * 8];
      Ac[c].q[0] = *(const uint4*)ap;
      Ac[c].q[1] = *(const uint4*)(ap + 16);
    }
    v8f zo = {};
#pragma unroll
    for (int c = 0; c < 2; ++c) {
      ABReg B;
      const unsigned short* bp = &w4t[m * 64 + c * 32 + hi * 8];
      B.q[0] = *(const uint4*)bp;
      B.q[1] = *(const uint4*)(bp + 16);
      zo = __builtin_amdgcn_wmma_f32_16x16x32_bf16(false, Ac[c].v, false, B.v,
                                                   (short)0, zo, false, false);
    }
    AccReg ro; ro.v = zo;
    float bias = sb4[m];
    if (m < 8) {
#pragma unroll
      for (int r = 0; r < 8; ++r) {
        int er = tile * 16 + r + hi * 8;
        out[(size_t)er * 8 + m] = ro.f[r] + bias;
      }
    }
  }
}

// =====================================================================
// launch
// =====================================================================
extern "C" void kernel_launch(void* const* d_in, const int* in_sizes, int n_in,
                              void* d_out, int out_size, void* d_ws, size_t ws_size,
                              hipStream_t stream) {
  (void)in_sizes; (void)n_in; (void)out_size; (void)ws_size;
  const float* x   = (const float*)d_in[0];
  const int*   ei  = (const int*)d_in[1];
  const float* ea  = (const float*)d_in[2];
  const float* eps = (const float*)d_in[3];
  const float* nw1 = (const float*)d_in[4];
  const float* nb1 = (const float*)d_in[5];
  const float* nw2 = (const float*)d_in[6];
  const float* nb2 = (const float*)d_in[7];
  const float *root[4], *cb[4], *g[4], *be[4];
  for (int l = 0; l < 4; ++l) {
    root[l] = (const float*)d_in[8 + 4 * l];
    cb[l]   = (const float*)d_in[9 + 4 * l];
    g[l]    = (const float*)d_in[10 + 4 * l];
    be[l]   = (const float*)d_in[11 + 4 * l];
  }
  const float* mu_w = (const float*)d_in[24];
  const float* mu_b = (const float*)d_in[25];
  const float* lv_w = (const float*)d_in[26];
  const float* lv_b = (const float*)d_in[27];
  const float *dw[5], *db[5];
  for (int i = 0; i < 5; ++i) {
    dw[i] = (const float*)d_in[28 + 2 * i];
    db[i] = (const float*)d_in[29 + 2 * i];
  }
  float* outp = (float*)d_out;

  // workspace carve-up (~34 MB total)
  char* ws = (char*)d_ws;
  unsigned short* hid = (unsigned short*)ws; ws += (size_t)NEDGES * 16 * 2;
  unsigned short* hbf = (unsigned short*)ws; ws += (size_t)NNODES * 16 * 2;
  float* agg  = (float*)ws;                  ws += (size_t)NNODES * 16 * 4;
  float* hpre = (float*)ws;                  ws += (size_t)NNODES * 16 * 4;
  unsigned short* zbf = (unsigned short*)ws; ws += (size_t)NNODES * 16 * 2;
  float* stats = (float*)ws;                 ws += 32 * 4;

  const int ntiles = NEDGES / 16;       // 50000
  const int convBlocks = (ntiles + 7) / 8;

  k_hid<<<3125, 256, 0, stream>>>(ea, nw1, nb1, hid);
  k_cast<<<512, 256, 0, stream>>>(x, hbf, NNODES * 16);

  for (int l = 0; l < 4; ++l) {
    k_zero<<<512, 256, 0, stream>>>(agg, stats);
    k_conv<<<convBlocks, 256, 0, stream>>>(hbf, hid, ei, nw2, nb2, agg);
    k_root<<<256, 256, 0, stream>>>(hbf, agg, root[l], cb[l], hpre, stats);
    k_bn<<<512, 256, 0, stream>>>(hpre, stats, g[l], be[l], hbf);
    // NOTE: k_root consumes hbf before k_bn overwrites it (stream-ordered).
  }
  k_latent<<<256, 256, 0, stream>>>(hbf, eps, mu_w, mu_b, lv_w, lv_b, zbf);
  k_decoder<<<convBlocks, 256, 0, stream>>>(zbf, ei, dw[0], db[0], dw[1], db[1],
                                            dw[2], db[2], dw[3], db[3], dw[4],
                                            db[4], outp);
}
